// Block_86663850099252
// MI455X (gfx1250) — matrix operand order repaired and checked by reading, not compiled, not run on hardware
//
#include <hip/hip_runtime.h>
#include <cstdint>

typedef unsigned short u16;
typedef __attribute__((ext_vector_type(16))) __bf16 v16bf;
typedef __attribute__((ext_vector_type(8)))  float  v8f;

enum { B_ = 8, T_ = 2048, C_ = 128, H_ = 4, HS_ = 32, R_ = B_ * T_, F_ = 4 * C_ };

// ---------- helpers ----------

__device__ __forceinline__ u16 f2bf(float f) {          // f32 -> bf16 RNE
  unsigned u = __float_as_uint(f);
  u += 0x7fffu + ((u >> 16) & 1u);
  return (u16)(u >> 16);
}

// A-fragment (16x32 bf16, row-major source): lane L holds row m=L&15,
// K chunks [8*hi, 8*hi+8) and [16+8*hi, 24+8*hi)  (hi = L>>4)
__device__ __forceinline__ void load_frag_a(v16bf& a, const u16* p) {
  ((uint4*)&a)[0] = *(const uint4*)p;
  ((uint4*)&a)[1] = *(const uint4*)(p + 16);
}
// B-fragment (32x16 bf16, from B^T [N,K] row-major): lane L holds col n=L&15,
// K chunks [8*hi, 8*hi+8) and [16+8*hi, 24+8*hi)
__device__ __forceinline__ void load_frag_b(v16bf& b, const u16* p) {
  ((uint4*)&b)[0] = *(const uint4*)p;
  ((uint4*)&b)[1] = *(const uint4*)(p + 16);
}

// 16x(16*NT) output strip, K = 32*KT.  A: [M,K] bf16 row-major.  BT: [N,K] bf16.
// One A-fragment per K-step is reused across all NT B-fragments.
template <int KT, int NT>
__device__ __forceinline__ void gemm_strip(const u16* __restrict__ A, int lda,
                                           const u16* __restrict__ BT, int ldb,
                                           int row0, int col0, int lane, v8f* acc) {
  const int mr = lane & 15, hi = lane >> 4;
  const u16* pa = A  + (size_t)(row0 + mr) * lda + hi * 8;
  const u16* pb = BT + (size_t)(col0 + mr) * ldb + hi * 8;
#pragma unroll
  for (int k = 0; k < KT; ++k) {
    v16bf a;
    load_frag_a(a, pa);
#pragma unroll
    for (int j = 0; j < NT; ++j) {
      v16bf b;
      load_frag_b(b, pb + (size_t)j * 16 * ldb);
      acc[j] = __builtin_amdgcn_wmma_f32_16x16x32_bf16(false, a, false, b,
                                                       (short)0, acc[j], false, false);
    }
    pa += 32; pb += 32;
  }
}

// ---------- weight conversion ----------

// Wq/Wk/Wv (H,C,HS) -> WqkvT [384][128] bf16, row n = which*128 + h*32 + d
__global__ void cvt_qkv_kernel(const float* __restrict__ Wq, const float* __restrict__ Wk,
                               const float* __restrict__ Wv, u16* __restrict__ dst) {
  int i = blockIdx.x * blockDim.x + threadIdx.x;
  if (i >= 384 * 128) return;
  int n = i >> 7, c = i & 127;
  int which = n >> 7, hd = n & 127, h = hd >> 5, d = hd & 31;
  const float* W = (which == 0) ? Wq : (which == 1) ? Wk : Wv;
  dst[i] = f2bf(W[h * (C_ * HS_) + c * HS_ + d]);
}

// src [K,N] f32 row-major -> dst [N,K] bf16
__global__ void cvtT_kernel(const float* __restrict__ src, u16* __restrict__ dst,
                            int K, int N) {
  int i = blockIdx.x * blockDim.x + threadIdx.x;
  if (i >= K * N) return;
  int n = i / K, k = i - n * K;
  dst[i] = f2bf(src[(size_t)k * N + n]);
}

// ---------- layernorm (one wave per 128-wide row) ----------

__global__ void ln_kernel(const float* __restrict__ x, const float* __restrict__ g,
                          const float* __restrict__ b, u16* __restrict__ out) {
  int lane = threadIdx.x & 31;
  int row  = (blockIdx.x * blockDim.x + threadIdx.x) >> 5;
  const float* xr = x + (size_t)row * C_;
  float4 v = *(const float4*)(xr + lane * 4);
  float s = v.x + v.y + v.z + v.w;
#pragma unroll
  for (int m = 16; m >= 1; m >>= 1) s += __shfl_xor(s, m, 32);
  float mu = s * (1.0f / C_);
  float dx = v.x - mu, dy = v.y - mu, dz = v.z - mu, dw = v.w - mu;
  float q = dx * dx + dy * dy + dz * dz + dw * dw;
#pragma unroll
  for (int m = 16; m >= 1; m >>= 1) q += __shfl_xor(q, m, 32);
  float rs = rsqrtf(q * (1.0f / C_) + 1e-5f);
  int c = lane * 4;
  float4 gv = *(const float4*)(g + c);
  float4 bv = *(const float4*)(b + c);
  u16* o = out + (size_t)row * C_ + c;
  o[0] = f2bf(dx * rs * gv.x + bv.x);
  o[1] = f2bf(dy * rs * gv.y + bv.y);
  o[2] = f2bf(dz * rs * gv.z + bv.z);
  o[3] = f2bf(dw * rs * gv.w + bv.w);
}

// ---------- QKV projection: h1[16384,128] x WqkvT[384,128] ----------
// Q: [B*H, T, 32] (scaled by HS^-0.5), K: [B*H, T, 32], V stored transposed [B*H, 32, T]
__global__ void qkv_kernel(const u16* __restrict__ h1, const u16* __restrict__ WqkvT,
                           u16* __restrict__ Q, u16* __restrict__ Kb, u16* __restrict__ VT) {
  int lane = threadIdx.x & 31;
  int tile = (blockIdx.x * blockDim.x + threadIdx.x) >> 5;   // 1024 * 6 strips
  int nt = tile % 6, mt = tile / 6;                          // strip = 16x64
  v8f acc[4] = {};
  gemm_strip<4, 4>(h1, C_, WqkvT, C_, mt * 16, nt * 64, lane, acc);
  int mr = lane & 15, hi = lane >> 4;
#pragma unroll
  for (int j = 0; j < 4; ++j) {
    int n = nt * 64 + j * 16 + mr;
    int which = n >> 7, hd = n & 127, h = hd >> 5, d = hd & 31;
#pragma unroll
    for (int r = 0; r < 8; ++r) {
      int row = mt * 16 + r + 8 * hi;          // token index
      int b = row >> 11, t = row & (T_ - 1);
      int bh = b * H_ + h;
      float v = acc[j][r];
      if (which == 0)
        Q[((size_t)bh * T_ + t) * HS_ + d] = f2bf(v * 0.17677669529663687f); // 1/sqrt(32)
      else if (which == 1)
        Kb[((size_t)bh * T_ + t) * HS_ + d] = f2bf(v);
      else
        VT[((size_t)bh * HS_ + d) * T_ + t] = f2bf(v);
    }
  }
}

// ---------- flash attention: one wave per (b,h,qtile of 16) ----------
__global__ void attn_kernel(const u16* __restrict__ Q, const u16* __restrict__ Kb,
                            const u16* __restrict__ VT, u16* __restrict__ Y) {
  const int lane = threadIdx.x & 31;
  const int wave = threadIdx.x >> 5;
  int gw = blockIdx.x * 8 + wave;            // 0..4095
  int qt = gw >> 5;                          // same for all waves in block
  int bh = gw & 31;
  int h = bh & (H_ - 1), b = bh >> 2;
  const u16* Qp = Q  + (size_t)bh * T_ * HS_;
  const u16* Kp = Kb + (size_t)bh * T_ * HS_;
  const u16* Vp = VT + (size_t)bh * HS_ * T_;   // [32][2048]
  const int mr = lane & 15, hi = lane >> 4;

  __shared__ u16 lds[8][16][32];
  u16 (*pt)[32] = lds[wave];

  v16bf qf;
  load_frag_a(qf, Qp + (size_t)(qt * 16 + mr) * HS_ + hi * 8);

  v8f o0 = {}, o1 = {};
  float mrow[8], lrow[8];
#pragma unroll
  for (int r = 0; r < 8; ++r) { mrow[r] = -1e30f; lrow[r] = 0.0f; }

  const int nk = qt * 16 + 16;               // keys to visit (causal)
  for (int kb = 0; kb < nk; kb += 32) {
    v16bf kf0, kf1;
    load_frag_b(kf0, Kp + (size_t)(kb + mr) * HS_ + hi * 8);
    load_frag_b(kf1, Kp + (size_t)(kb + 16 + mr) * HS_ + hi * 8);
    v8f z0 = {}, z1 = {};
    v8f s0 = __builtin_amdgcn_wmma_f32_16x16x32_bf16(false, qf, false, kf0,
                                                     (short)0, z0, false, false);
    v8f s1 = __builtin_amdgcn_wmma_f32_16x16x32_bf16(false, qf, false, kf1,
                                                     (short)0, z1, false, false);
    if (kb + 32 > qt * 16) {                 // tile touches/crosses diagonal
      int k0 = kb + mr, k1 = k0 + 16;
#pragma unroll
      for (int r = 0; r < 8; ++r) {
        int qi = qt * 16 + r + 8 * hi;
        if (k0 > qi) s0[r] = -1e30f;
        if (k1 > qi) s1[r] = -1e30f;
      }
    }
#pragma unroll
    for (int r = 0; r < 8; ++r) {
      float mx = fmaxf(s0[r], s1[r]);
      mx = fmaxf(mx, __shfl_xor(mx, 1, 32));
      mx = fmaxf(mx, __shfl_xor(mx, 2, 32));
      mx = fmaxf(mx, __shfl_xor(mx, 4, 32));
      mx = fmaxf(mx, __shfl_xor(mx, 8, 32));
      float mn = fmaxf(mrow[r], mx);
      float sc = __expf(mrow[r] - mn);
      float p0 = __expf(s0[r] - mn);
      float p1 = __expf(s1[r] - mn);
      float rs = p0 + p1;
      rs += __shfl_xor(rs, 1, 32);
      rs += __shfl_xor(rs, 2, 32);
      rs += __shfl_xor(rs, 4, 32);
      rs += __shfl_xor(rs, 8, 32);
      lrow[r] = lrow[r] * sc + rs;
      mrow[r] = mn;
      o0[r] *= sc;
      o1[r] *= sc;
      pt[r + 8 * hi][mr]      = f2bf(p0);    // P in (m,n) order for A-frag reload
      pt[r + 8 * hi][mr + 16] = f2bf(p1);
    }
    asm volatile("s_wait_dscnt 0" ::: "memory");   // order ds_store -> ds_load
    v16bf pf;
    load_frag_a(pf, &pt[mr][hi * 8]);
    v16bf vf0, vf1;
    load_frag_b(vf0, Vp + (size_t)mr * T_ + kb + hi * 8);
    load_frag_b(vf1, Vp + (size_t)(mr + 16) * T_ + kb + hi * 8);
    o0 = __builtin_amdgcn_wmma_f32_16x16x32_bf16(false, pf, false, vf0,
                                                 (short)0, o0, false, false);
    o1 = __builtin_amdgcn_wmma_f32_16x16x32_bf16(false, pf, false, vf1,
                                                 (short)0, o1, false, false);
  }
#pragma unroll
  for (int r = 0; r < 8; ++r) {
    float inv = 1.0f / lrow[r];
    int t = qt * 16 + r + 8 * hi;
    size_t row = ((size_t)b * T_ + t) * C_ + h * HS_;
    Y[row + mr]      = f2bf(o0[r] * inv);
    Y[row + 16 + mr] = f2bf(o1[r] * inv);
  }
}

// ---------- output projection + residual: X1 = x + Y@Wp + bp ----------
__global__ void proj_kernel(const u16* __restrict__ Y, const u16* __restrict__ WpT,
                            const float* __restrict__ x, const float* __restrict__ bp,
                            float* __restrict__ X1) {
  int lane = threadIdx.x & 31;
  int tile = (blockIdx.x * blockDim.x + threadIdx.x) >> 5;   // 1024 * 2 strips
  int nt = tile & 1, mt = tile >> 1;
  v8f acc[4] = {};
  gemm_strip<4, 4>(Y, C_, WpT, C_, mt * 16, nt * 64, lane, acc);
  int mr = lane & 15, hi = lane >> 4;
#pragma unroll
  for (int j = 0; j < 4; ++j) {
    int n = nt * 64 + j * 16 + mr;
    float bias = bp[n];
#pragma unroll
    for (int r = 0; r < 8; ++r) {
      size_t idx = (size_t)(mt * 16 + r + 8 * hi) * C_ + n;
      X1[idx] = x[idx] + acc[j][r] + bias;
    }
  }
}

// ---------- MLP fc1 + exact GELU: U = gelu(h2 @ W1) ----------
__global__ void mlp1_kernel(const u16* __restrict__ h2, const u16* __restrict__ W1T,
                            u16* __restrict__ U) {
  int lane = threadIdx.x & 31;
  int tile = (blockIdx.x * blockDim.x + threadIdx.x) >> 5;   // 1024 * 8 strips
  int nt = tile & 7, mt = tile >> 3;
  v8f acc[4] = {};
  gemm_strip<4, 4>(h2, C_, W1T, C_, mt * 16, nt * 64, lane, acc);
  int mr = lane & 15, hi = lane >> 4;
#pragma unroll
  for (int j = 0; j < 4; ++j) {
    int n = nt * 64 + j * 16 + mr;
#pragma unroll
    for (int r = 0; r < 8; ++r) {
      float v = acc[j][r];
      float ge = 0.5f * v * (1.0f + erff(v * 0.7071067811865475f));
      U[(size_t)(mt * 16 + r + 8 * hi) * F_ + n] = f2bf(ge);
    }
  }
}

// ---------- MLP fc2 + residual to output: out = X1 + U @ W2 ----------
__global__ void mlp2_kernel(const u16* __restrict__ U, const u16* __restrict__ W2T,
                            const float* __restrict__ X1, float* __restrict__ out) {
  int lane = threadIdx.x & 31;
  int tile = (blockIdx.x * blockDim.x + threadIdx.x) >> 5;   // 1024 * 2 strips
  int nt = tile & 1, mt = tile >> 1;
  v8f acc[4] = {};
  gemm_strip<16, 4>(U, F_, W2T, F_, mt * 16, nt * 64, lane, acc);
  int mr = lane & 15, hi = lane >> 4;
#pragma unroll
  for (int j = 0; j < 4; ++j) {
    int n = nt * 64 + j * 16 + mr;
#pragma unroll
    for (int r = 0; r < 8; ++r) {
      size_t idx = (size_t)(mt * 16 + r + 8 * hi) * C_ + n;
      out[idx] = X1[idx] + acc[j][r];
    }
  }
}

// ---------- launch ----------

extern "C" void kernel_launch(void* const* d_in, const int* in_sizes, int n_in,
                              void* d_out, int out_size, void* d_ws, size_t ws_size,
                              hipStream_t stream) {
  const float* x    = (const float*)d_in[0];
  const float* ln1g = (const float*)d_in[1];
  const float* ln1b = (const float*)d_in[2];
  const float* Wq   = (const float*)d_in[3];
  const float* Wk   = (const float*)d_in[4];
  const float* Wv   = (const float*)d_in[5];
  const float* Wp   = (const float*)d_in[6];
  const float* bp   = (const float*)d_in[7];
  const float* ln2g = (const float*)d_in[8];
  const float* ln2b = (const float*)d_in[9];
  const float* W1   = (const float*)d_in[10];
  const float* W2   = (const float*)d_in[11];
  float* out = (float*)d_out;

  char* ws = (char*)d_ws;
  size_t off = 0;
  auto carve = [&](size_t bytes) -> char* {
    char* p = ws + off;
    off += (bytes + 255) & ~(size_t)255;
    return p;
  };
  u16*   h1    = (u16*)carve((size_t)R_ * C_ * 2);
  u16*   h2    = (u16*)carve((size_t)R_ * C_ * 2);
  u16*   WqkvT = (u16*)carve((size_t)384 * C_ * 2);
  u16*   WpT   = (u16*)carve((size_t)C_ * C_ * 2);
  u16*   W1T   = (u16*)carve((size_t)F_ * C_ * 2);
  u16*   W2T   = (u16*)carve((size_t)C_ * F_ * 2);
  u16*   Qb    = (u16*)carve((size_t)B_ * H_ * T_ * HS_ * 2);
  u16*   Kb    = (u16*)carve((size_t)B_ * H_ * T_ * HS_ * 2);
  u16*   VT    = (u16*)carve((size_t)B_ * H_ * T_ * HS_ * 2);
  u16*   Y     = (u16*)carve((size_t)R_ * C_ * 2);
  float* X1    = (float*)carve((size_t)R_ * C_ * 4);
  u16*   U     = (u16*)carve((size_t)R_ * F_ * 2);
  (void)in_sizes; (void)n_in; (void)out_size; (void)ws_size;

  // weight conversion (bf16, pre-transposed)
  cvt_qkv_kernel<<<(384 * 128 + 255) / 256, 256, 0, stream>>>(Wq, Wk, Wv, WqkvT);
  cvtT_kernel<<<(C_ * C_ + 255) / 256, 256, 0, stream>>>(Wp, WpT, C_, C_);
  cvtT_kernel<<<(C_ * F_ + 255) / 256, 256, 0, stream>>>(W1, W1T, C_, F_);
  cvtT_kernel<<<(F_ * C_ + 255) / 256, 256, 0, stream>>>(W2, W2T, F_, C_);

  // LN1 -> h1
  ln_kernel<<<R_ / 8, 256, 0, stream>>>(x, ln1g, ln1b, h1);
  // QKV projection (16x64 strips)
  qkv_kernel<<<(R_ / 16) * 6 / 8, 256, 0, stream>>>(h1, WqkvT, Qb, Kb, VT);
  // attention -> Y (bf16, heads concatenated)
  attn_kernel<<<(B_ * H_ * (T_ / 16)) / 8, 256, 0, stream>>>(Qb, Kb, VT, Y);
  // output projection + residual -> X1 (f32)
  proj_kernel<<<(R_ / 16) * 2 / 8, 256, 0, stream>>>(Y, WpT, x, bp, X1);
  // LN2 -> h2
  ln_kernel<<<R_ / 8, 256, 0, stream>>>(X1, ln2g, ln2b, h2);
  // MLP
  mlp1_kernel<<<(R_ / 16) * 8 / 8, 256, 0, stream>>>(h2, W1T, U);
  mlp2_kernel<<<(R_ / 16) * 2 / 8, 256, 0, stream>>>(U, W2T, X1, out);
}
